// LSTMModel_24704651886777
// MI455X (gfx1250) — matrix operand order repaired and checked by reading, not compiled, not run on hardware
//
#include <hip/hip_runtime.h>
#include <hip/hip_bf16.h>
#include <cmath>

typedef __attribute__((ext_vector_type(16))) _Float16 v16h;
typedef __attribute__((ext_vector_type(8)))  float    v8f;

#define BB 256
#define TT 512

// ---- fast transcendentals on the recurrence critical path -----------------
#if defined(__has_builtin)
#  if __has_builtin(__builtin_amdgcn_exp2f)
#    define EXP2F(x) __builtin_amdgcn_exp2f(x)
#  endif
#  if __has_builtin(__builtin_amdgcn_rcpf)
#    define RCPF(x) __builtin_amdgcn_rcpf(x)
#  endif
#  if __has_builtin(__builtin_amdgcn_tanhf)
#    define TANHF_HW 1
#  endif
#endif
#ifndef EXP2F
#  define EXP2F(x) exp2f(x)
#endif
#ifndef RCPF
#  define RCPF(x) (1.0f / (x))
#endif

__device__ __forceinline__ float fast_sigmoid(float x) {
    // sigma(x) = rcp(1 + 2^(-x*log2 e))
    return RCPF(1.0f + EXP2F(x * -1.4426950408889634f));
}
__device__ __forceinline__ float fast_tanh(float x) {
#ifdef TANHF_HW
    return __builtin_amdgcn_tanhf(x);   // v_tanh_f32
#else
    // tanh(x) = 2*sigma(2x) - 1 = 2*rcp(1 + 2^(-2x*log2 e)) - 1
    const float t = EXP2F(x * -2.8853900817779268f);
    return __builtin_fmaf(2.0f, RCPF(1.0f + t), -1.0f);
#endif
}

// ---------------------------------------------------------------------------
// Pack fp32 (wih[4H,I], whh[4H,H]) into f16 Wt[KP][4H] (transposed, K-major,
// zero padded to KP) and bias[4H] = bih + bhh.
// ---------------------------------------------------------------------------
__global__ void pack_weights_kernel(const float* __restrict__ wih,
                                    const float* __restrict__ whh,
                                    const float* __restrict__ bih,
                                    const float* __restrict__ bhh,
                                    _Float16* __restrict__ Wt,
                                    float* __restrict__ bias,
                                    int I, int H, int KP) {
    const int NG = 4 * H;
    const int total = KP * NG;
    for (int idx = blockIdx.x * blockDim.x + threadIdx.x; idx < total;
         idx += gridDim.x * blockDim.x) {
        int k = idx / NG;
        int n = idx - k * NG;
        float v = 0.f;
        if (k < I)            v = wih[n * I + k];
        else if (k < I + H)   v = whh[n * H + (k - I)];
        Wt[idx] = (_Float16)v;
        if (idx < NG) bias[idx] = bih[idx] + bhh[idx];
    }
}

// ---------------------------------------------------------------------------
// One LSTM layer, full time recurrence.
//   I = input size, H = hidden size, KP = roundup32(I+H)
// Workgroup: 16 batch rows; NW = 4H/16 waves, one 16-col gate tile per wave.
// Weight B-fragments live in VGPRs for the whole t-loop; h/c live in LDS.
// x_t is software-pipelined through registers (loaded one step ahead).
// Two barriers per step.
// ---------------------------------------------------------------------------
template<int I, int H, int KP>
__global__ void __launch_bounds__((4 * H / 16) * 32)
lstm_layer_kernel(const float* __restrict__ xin,      // [B,T,I]
                  const _Float16* __restrict__ Wt,    // [KP][4H] f16
                  const float* __restrict__ bias,     // [4H]
                  float* __restrict__ hout) {         // [B,T,H]
    constexpr int NG   = 4 * H;
    constexpr int NW   = NG / 16;
    constexpr int KT   = KP / 32;
    constexpr int NTHR = NW * 32;                 // == 8*H
    constexpr int XPT  = (16 * I + NTHR - 1) / NTHR;  // x elems / thread

    __shared__ _Float16 aLds[16][KP];   // [x_t | h_{t-1} | 0-pad], f16
    __shared__ float    gLds[16][NG];   // pre-activation gates
    __shared__ float    cLds[16][H];    // cell state
    __shared__ float    bLds[NG];

    const int tid  = threadIdx.x;
    const int lane = tid & 31;
    const int wave = tid >> 5;
    const int b0   = blockIdx.x * 16;
    const bool hi  = (lane >= 16);

    // ---- init state (h0 = c0 = 0, zero K-padding) -------------------------
    for (int idx = tid; idx < 16 * KP; idx += NTHR)
        (&aLds[0][0])[idx] = (_Float16)0.f;
    for (int idx = tid; idx < 16 * H; idx += NTHR)
        (&cLds[0][0])[idx] = 0.f;
    for (int idx = tid; idx < NG; idx += NTHR)
        bLds[idx] = bias[idx];

    // ---- loop-invariant weight B-fragments in VGPRs -----------------------
    // B (32x16) f16, k order used here (matches A): lanes 0-15 K=0..15, lanes 16-31 K=16..31, N=lane&15
    const int ncol = wave * 16 + (lane & 15);
    v16h bfrag[KT];
#pragma unroll
    for (int kt = 0; kt < KT; ++kt) {
#pragma unroll
        for (int e = 0; e < 16; ++e) {
            const int K = kt * 32 + e + (hi ? 16 : 0);
            bfrag[kt][e] = Wt[K * NG + ncol];
        }
    }

    // ---- fixed per-thread mappings ----------------------------------------
    const int arow  = lane & 15;
    const int rbase = hi ? 8 : 0;

    // elementwise: exactly 2 elements per thread (16*H == 2*NTHR)
    const int e0 = tid, e1 = tid + NTHR;
    const int r0 = e0 / H, j0 = e0 - r0 * H;
    const int r1 = e1 / H, j1 = e1 - r1 * H;
    float* hp0 = hout + (size_t)(b0 + r0) * TT * H + j0;
    float* hp1 = hout + (size_t)(b0 + r1) * TT * H + j1;

    // x streaming: per-thread source pointers + LDS targets
    const float* xp[XPT];
    _Float16*    axp[XPT];
    bool         xv[XPT];
    float        xreg[XPT];
#pragma unroll
    for (int u = 0; u < XPT; ++u) {
        const int idx = tid + u * NTHR;
        xv[u] = (idx < 16 * I);
        const int row = xv[u] ? idx / I : 0;
        const int i   = xv[u] ? idx - row * I : 0;
        xp[u]  = xin + (size_t)(b0 + row) * TT * I + i;
        axp[u] = &aLds[row][i];
        if (xv[u]) xreg[u] = xp[u][0];   // prefetch t = 0
    }
    __syncthreads();

    const float bcol = bLds[ncol];

    for (int t = 0; t < TT; ++t) {
        // commit x_t (prefetched last iteration) to LDS
#pragma unroll
        for (int u = 0; u < XPT; ++u)
            if (xv[u]) axp[u][0] = (_Float16)xreg[u];

        // prefetch x_{t+1} (clamped at the end; value unused then)
        const int tn = (t + 1 < TT) ? (t + 1) : t;
#pragma unroll
        for (int u = 0; u < XPT; ++u)
            if (xv[u]) xreg[u] = xp[u][(size_t)tn * I];

        __syncthreads();    // A = [x_t | h_{t-1}] ready

        v8f acc;
#pragma unroll
        for (int j = 0; j < 8; ++j) acc[j] = bcol;

#pragma unroll
        for (int kt = 0; kt < KT; ++kt) {
            // A (16x32) f16, k order used here (matches B): lanes 0-15 K=0..15; lanes 16-31 K=16..31
            v16h a;
            const int kbase = kt * 32 + (hi ? 16 : 0);
#pragma unroll
            for (int e = 0; e < 16; ++e) {
                const int K = kbase + e;
                a[e] = aLds[arow][K];
            }
            acc = __builtin_amdgcn_wmma_f32_16x16x32_f16(
                false, a, false, bfrag[kt], (short)0, acc, false, false);
        }

        // C/D layout: VGPR j -> row j + (hi?8:0), col = ncol
#pragma unroll
        for (int j = 0; j < 8; ++j)
            gLds[rbase + j][ncol] = acc[j];

        __syncthreads();    // gates ready

        // elementwise LSTM cell (gate order i,f,g,o), 2 elems/thread
        {
            const float ig = fast_sigmoid(gLds[r0][j0]);
            const float fg = fast_sigmoid(gLds[r0][H + j0]);
            const float gg = fast_tanh(gLds[r0][2 * H + j0]);
            const float og = fast_sigmoid(gLds[r0][3 * H + j0]);
            const float c  = fg * cLds[r0][j0] + ig * gg;
            cLds[r0][j0] = c;
            const float hv = og * fast_tanh(c);
            aLds[r0][I + j0] = (_Float16)hv;
            hp0[(size_t)t * H] = hv;
        }
        {
            const float ig = fast_sigmoid(gLds[r1][j1]);
            const float fg = fast_sigmoid(gLds[r1][H + j1]);
            const float gg = fast_tanh(gLds[r1][2 * H + j1]);
            const float og = fast_sigmoid(gLds[r1][3 * H + j1]);
            const float c  = fg * cLds[r1][j1] + ig * gg;
            cLds[r1][j1] = c;
            const float hv = og * fast_tanh(c);
            aLds[r1][I + j1] = (_Float16)hv;
            hp1[(size_t)t * H] = hv;
        }
        // no barrier here: h-writes are fenced by next iteration's pre-WMMA
        // barrier; x-region writes don't conflict with anything in flight.
    }
}

// ---------------------------------------------------------------------------
// Final FC: out[b,t] = h[b,t,:] . fc_w + fc_b   (H = 32)
// ---------------------------------------------------------------------------
__global__ void fc_kernel(const float* __restrict__ hin,
                          const float* __restrict__ fcw,
                          const float* __restrict__ fcb,
                          float* __restrict__ out, int n) {
    const int idx = blockIdx.x * blockDim.x + threadIdx.x;
    if (idx >= n) return;
    float s = fcb[0];
#pragma unroll
    for (int j = 0; j < 32; ++j)
        s += hin[(size_t)idx * 32 + j] * fcw[j];
    out[idx] = s;
}

// ---------------------------------------------------------------------------
extern "C" void kernel_launch(void* const* d_in, const int* in_sizes, int n_in,
                              void* d_out, int out_size, void* d_ws, size_t ws_size,
                              hipStream_t stream) {
    (void)in_sizes; (void)n_in; (void)out_size; (void)ws_size;

    struct LD { int I, H, KP; };
    static const LD dims[12] = {
        {1, 32, 64}, {32, 32, 64}, {32, 32, 64},       // lstm1
        {32, 64, 96}, {64, 64, 128}, {64, 64, 128},    // lstm21
        {64, 64, 128}, {64, 64, 128}, {64, 64, 128},   // lstm22
        {64, 32, 96}, {32, 32, 64}, {32, 32, 64}       // lstm3
    };

    char* ws = (char*)d_ws;
    float* act0 = (float*)ws;
    float* act1 = (float*)(ws + (size_t)BB * TT * 64 * sizeof(float));
    char*  wbase = ws + 2 * (size_t)BB * TT * 64 * sizeof(float);
    const size_t WT_BYTES = (size_t)128 * 256 * sizeof(_Float16);  // max KP*4H
    const size_t WSTRIDE  = WT_BYTES + 256 * sizeof(float);        // + bias

    // ---- pack all 12 layers' weights --------------------------------------
    for (int L = 0; L < 12; ++L) {
        const float* wih = (const float*)d_in[1 + L * 4 + 0];
        const float* whh = (const float*)d_in[1 + L * 4 + 1];
        const float* bih = (const float*)d_in[1 + L * 4 + 2];
        const float* bhh = (const float*)d_in[1 + L * 4 + 3];
        _Float16* Wtp = (_Float16*)(wbase + L * WSTRIDE);
        float*    bia = (float*)(wbase + L * WSTRIDE + WT_BYTES);
        const int total = dims[L].KP * 4 * dims[L].H;
        pack_weights_kernel<<<(total + 255) / 256, 256, 0, stream>>>(
            wih, whh, bih, bhh, Wtp, bia, dims[L].I, dims[L].H, dims[L].KP);
    }

    // ---- 12 recurrent layers, ping-pong activations -----------------------
    const float* cur = (const float*)d_in[0];   // x [B,T,1]
    float* bufs[2] = {act0, act1};
    int pb = 0;
    for (int L = 0; L < 12; ++L) {
        const _Float16* Wtp = (const _Float16*)(wbase + L * WSTRIDE);
        const float*    bia = (const float*)(wbase + L * WSTRIDE + WT_BYTES);
        float* outL = bufs[pb];
        const dim3 grid(BB / 16);
        const int I = dims[L].I, H = dims[L].H, KP = dims[L].KP;
        if (I == 1 && H == 32 && KP == 64)
            lstm_layer_kernel<1, 32, 64><<<grid, (4*32/16)*32, 0, stream>>>(cur, Wtp, bia, outL);
        else if (I == 32 && H == 32 && KP == 64)
            lstm_layer_kernel<32, 32, 64><<<grid, (4*32/16)*32, 0, stream>>>(cur, Wtp, bia, outL);
        else if (I == 32 && H == 64 && KP == 96)
            lstm_layer_kernel<32, 64, 96><<<grid, (4*64/16)*32, 0, stream>>>(cur, Wtp, bia, outL);
        else if (I == 64 && H == 64 && KP == 128)
            lstm_layer_kernel<64, 64, 128><<<grid, (4*64/16)*32, 0, stream>>>(cur, Wtp, bia, outL);
        else /* I==64 && H==32 && KP==96 */
            lstm_layer_kernel<64, 32, 96><<<grid, (4*32/16)*32, 0, stream>>>(cur, Wtp, bia, outL);
        cur = outL;
        pb ^= 1;
    }

    // ---- final FC ---------------------------------------------------------
    const float* fcw = (const float*)d_in[49];
    const float* fcb = (const float*)d_in[50];
    fc_kernel<<<(BB * TT + 255) / 256, 256, 0, stream>>>(
        cur, fcw, fcb, (float*)d_out, BB * TT);
}
